// ConditionedDeepKoopmanTransition_30305289240696
// MI455X (gfx1250) — matrix operand order repaired and checked by reading, not compiled, not run on hardware
//
#include <hip/hip_runtime.h>

// ---------------------------------------------------------------------------
// ConditionedDeepKoopmanTransition for MI455X (gfx1250, wave32, WMMA bf16)
//
// Key optimization: never materialize Kt (B,128,128) = 268MB. Instead:
//   z_next = M @ A^T  where M[b,(j,m)] = feat[b,j] * hzx[b,m]   (on-the-fly)
//            A[i,(j,m)] = packed {WK,bK | WL,bL}  (8.25MB bf16, L2-resident)
// hzx = [hz(192), 1.0, zero-pad] padded to 224 (=7*32) so biases fold in and
// the K loop is clean multiples of the 16x16x32 bf16 WMMA K dimension.
// ---------------------------------------------------------------------------

#define BATCH 4096
#define DYN   128
#define STAT  64
#define UDIM  16
#define NOBS  50
#define HID   256
#define HZ    192
#define HZP   224               // hz padded (192 data + 1 bias + 31 zero)
#define SELP  224               // selector input 193 padded to 224
#define KBIG  (DYN*HZP + UDIM*HZP)   // 28672 + 3584 = 32256
#define KOFFU (DYN*HZP)              // 28672

typedef __bf16 v16bf __attribute__((ext_vector_type(16)));
typedef float  v8f   __attribute__((ext_vector_type(8)));

union ABu { v16bf v; uint4 q[2]; };

static __device__ __forceinline__ unsigned short f2bf(float f) {
    union { float f; unsigned int u; } c; c.f = f;
    unsigned int u = c.u;
    return (unsigned short)((u + 0x7FFFu + ((u >> 16) & 1u)) >> 16);  // RNE
}

// ---------------------------------------------------------------------------
// Packing kernels
// ---------------------------------------------------------------------------

// x = [z_dyn | z_static | dt | 0-pad] -> bf16 (B x 224)
__global__ void k_pack_x(const float* __restrict__ zd, const float* __restrict__ zs,
                         const float* __restrict__ dt, unsigned short* __restrict__ xb) {
    int idx = blockIdx.x * 256 + threadIdx.x;
    if (idx >= BATCH * SELP) return;
    int b = idx / SELP, c = idx % SELP;
    float v = 0.f;
    if (c < DYN)            v = zd[b * DYN + c];
    else if (c < DYN + STAT) v = zs[b * STAT + (c - DYN)];
    else if (c == DYN + STAT) v = dt[b];
    xb[idx] = f2bf(v);
}

// zero-padded bf16 weight pack: W (N x K) -> Wp (N x Kp)
__global__ void k_pack_w(const float* __restrict__ W, unsigned short* __restrict__ Wp,
                         int K, int Kp, int total) {
    int idx = blockIdx.x * 256 + threadIdx.x;
    if (idx >= total) return;
    int n = idx / Kp, k = idx % Kp;
    Wp[idx] = f2bf(k < K ? W[n * K + k] : 0.f);
}

// Big packed weight: A (128 x 32256) bf16.
//   k < 28672:  j=k/224, m=k%224: m<192 -> WK[(i*128+j)*192+m]; m==192 -> bK[i*128+j]
//   else t=k-28672: u=t/224, m=t%224: m<192 -> WL[(i*16+u)*192+m]; m==192 -> bL[i*16+u]
__global__ void k_pack_A(const float* __restrict__ WK, const float* __restrict__ bK,
                         const float* __restrict__ WL, const float* __restrict__ bL,
                         unsigned short* __restrict__ A) {
    int idx = blockIdx.x * 256 + threadIdx.x;
    if (idx >= DYN * KBIG) return;
    int i = idx / KBIG, k = idx % KBIG;
    float v = 0.f;
    if (k < KOFFU) {
        int j = k / HZP, m = k % HZP;
        if (m < HZ)       v = WK[(i * DYN + j) * HZ + m];
        else if (m == HZ) v = bK[i * DYN + j];
    } else {
        int t = k - KOFFU;
        int u = t / HZP, m = t % HZP;
        if (m < HZ)       v = WL[(i * UDIM + u) * HZ + m];
        else if (m == HZ) v = bL[i * UDIM + u];
    }
    A[idx] = f2bf(v);
}

// ut_dt = ut * dt  (f32, B x 16)
__global__ void k_utdt(const float* __restrict__ ut, const float* __restrict__ dt,
                       float* __restrict__ out) {
    int idx = blockIdx.x * 256 + threadIdx.x;
    if (idx >= BATCH * UDIM) return;
    out[idx] = ut[idx] * dt[idx / UDIM];
}

// hzx (f32, B x 224): [hz | 1.0 | zeros]
__global__ void k_hzx(const float* __restrict__ hz, float* __restrict__ hzx) {
    int idx = blockIdx.x * 256 + threadIdx.x;
    if (idx >= BATCH * HZP) return;
    int m = idx % HZP;
    float v = 0.f;
    if (m < HZ) v = hz[(idx / HZP) * HZ + m];
    else if (m == HZ) v = 1.f;
    hzx[idx] = v;
}

// ---------------------------------------------------------------------------
// Generic bf16 WMMA GEMM: C(B x N) = act(B x Kp) @ Wp^T(N x Kp) + bias
// One wave per 16x16 tile. Operand striping used here (wave32, contiguous k; A and B match):
//   A: lane row = lane%16; k = (lane/16)*16 + [0..15] contiguous
//   B: lane col = lane%16; k = (lane/16)*16 + [0..15] contiguous
// ---------------------------------------------------------------------------
__global__ __launch_bounds__(32) void k_gemm(const unsigned short* __restrict__ Act,
                                             const unsigned short* __restrict__ Wp,
                                             const float* __restrict__ bias,
                                             float* __restrict__ Cout, int Kp, int N) {
    int lane = threadIdx.x;
    int row  = blockIdx.x * 16 + (lane & 15);
    int ncol = blockIdx.y * 16 + (lane & 15);
    int ka   = (lane >> 4) * 16;
    int kb   = (lane >> 4) * 16;
    const unsigned short* arow = Act + row * Kp;
    const unsigned short* wrow = Wp + ncol * Kp;
    v8f acc = {};
    for (int k0 = 0; k0 < Kp; k0 += 32) {
        ABu a, b;
        a.q[0] = *(const uint4*)(arow + k0 + ka);
        a.q[1] = *(const uint4*)(arow + k0 + ka + 8);
        const uint4* pb = (const uint4*)(wrow + k0 + kb);
        b.q[0] = pb[0]; b.q[1] = pb[1];
        acc = __builtin_amdgcn_wmma_f32_16x16x32_bf16(false, a.v, false, b.v,
                                                      (short)0, acc, false, false);
    }
    float bv = bias ? bias[ncol] : 0.f;
    int orow = blockIdx.x * 16 + (lane >> 4) * 8;
    int ocol = ncol;
#pragma unroll
    for (int r = 0; r < 8; ++r) Cout[(orow + r) * N + ocol] = acc[r] + bv;
}

// BatchNorm training-mode stats -> scale/shift per column
__global__ __launch_bounds__(256) void k_bnstats(const float* __restrict__ pre,
                                                 const float* __restrict__ g,
                                                 const float* __restrict__ be,
                                                 float* __restrict__ scale,
                                                 float* __restrict__ shift, int N) {
    __shared__ float ss[256], sq[256];
    int col = blockIdx.x;
    float s = 0.f, q = 0.f;
    for (int r = threadIdx.x; r < BATCH; r += 256) {
        float v = pre[r * N + col]; s += v; q += v * v;
    }
    ss[threadIdx.x] = s; sq[threadIdx.x] = q; __syncthreads();
    for (int o = 128; o > 0; o >>= 1) {
        if (threadIdx.x < o) { ss[threadIdx.x] += ss[threadIdx.x + o];
                               sq[threadIdx.x] += sq[threadIdx.x + o]; }
        __syncthreads();
    }
    if (threadIdx.x == 0) {
        float m  = ss[0] / (float)BATCH;
        float vr = sq[0] / (float)BATCH - m * m;
        float sc = g[col] * rsqrtf(vr + 1e-5f);
        scale[col] = sc; shift[col] = be[col] - m * sc;
    }
}

// h = bf16(relu(pre*scale + shift))
__global__ void k_bnact(const float* __restrict__ pre, const float* __restrict__ scale,
                        const float* __restrict__ shift, unsigned short* __restrict__ h,
                        int N) {
    int idx = blockIdx.x * 256 + threadIdx.x;
    if (idx >= BATCH * N) return;
    int n = idx % N;
    float v = pre[idx] * scale[n] + shift[n];
    h[idx] = f2bf(v > 0.f ? v : 0.f);
}

// ---------------------------------------------------------------------------
// Fused Koopman transition GEMM:  z_next(4096 x 128) = M @ A^T
// WG = 256 threads (8 waves), 16 batch rows x 128 outputs, grid = 256.
// Per macro-step (144 = 128 z cols + 16 u cols): generate the 16x224 bf16
// M-tile into a double-buffered LDS tile (ONE barrier per step), then each
// wave runs 7 x v_wmma_f32_16x16x32_bf16 against the L2-resident packed
// weight, with global_prefetch of the next weight chunk issued up front.
// ---------------------------------------------------------------------------
__global__ __launch_bounds__(256) void k_koopman(const float* __restrict__ zdyn,
                                                 const float* __restrict__ utdt,
                                                 const float* __restrict__ hzx,
                                                 const unsigned short* __restrict__ Aw,
                                                 float* __restrict__ zout) {
    __shared__ float zs[16 * DYN];
    __shared__ float us[16 * UDIM];
    __shared__ float hs[16 * HZP];
    __shared__ alignas(16) unsigned short at[2][16 * HZP];   // double buffer

    int tid = threadIdx.x;
    int bbase = blockIdx.x * 16;
    for (int i = tid; i < 16 * DYN;  i += 256) zs[i] = zdyn[bbase * DYN + i];
    for (int i = tid; i < 16 * UDIM; i += 256) us[i] = utdt[bbase * UDIM + i];
    for (int i = tid; i < 16 * HZP;  i += 256) hs[i] = hzx[bbase * HZP + i];

    int lane = tid & 31;
    int wave = tid >> 5;                  // n-tile 0..7
    int genrow = tid & 15;                // generator: row
    int genm0  = (tid >> 4) * 14;         // generator: 14 m-values each
    int arow = lane & 15;
    int ka   = (lane >> 4) * 16;
    int kb   = (lane >> 4) * 16;
    const unsigned short* wrow = Aw + (size_t)(wave * 16 + (lane & 15)) * KBIG;

    __syncthreads();

    v8f acc = {};
    for (int src = 0; src < DYN + UDIM; ++src) {
        unsigned short* atw = &at[src & 1][0];
        float zv = (src < DYN) ? zs[genrow * DYN + src]
                               : us[genrow * UDIM + (src - DYN)];
#pragma unroll
        for (int e = 0; e < 14; ++e) {
            int m = genm0 + e;
            atw[genrow * HZP + m] = f2bf(zv * hs[genrow * HZP + m]);
        }
        int kbase = (src < DYN) ? src * HZP : KOFFU + (src - DYN) * HZP;
        // prefetch next macro-step's weight chunk (L2-resident stream)
        if (src + 1 < DYN + UDIM) {
            int knext = (src + 1 < DYN) ? (src + 1) * HZP
                                        : KOFFU + (src + 1 - DYN) * HZP;
            __builtin_prefetch(wrow + knext + kb, 0, 0);
        }
        __syncthreads();   // single barrier: double buffer protects gen vs use
        const unsigned short* atr = &at[src & 1][0];
#pragma unroll
        for (int mb = 0; mb < 7; ++mb) {
            ABu a, b;
            a.q[0] = *(const uint4*)(atr + arow * HZP + mb * 32 + ka);
            a.q[1] = *(const uint4*)(atr + arow * HZP + mb * 32 + ka + 8);
            const uint4* pb = (const uint4*)(wrow + kbase + mb * 32 + kb);
            b.q[0] = pb[0]; b.q[1] = pb[1];
            acc = __builtin_amdgcn_wmma_f32_16x16x32_bf16(false, a.v, false, b.v,
                                                          (short)0, acc, false, false);
        }
    }
    int orow = bbase + (lane >> 4) * 8;
    int ocol = wave * 16 + (lane & 15);
#pragma unroll
    for (int r = 0; r < 8; ++r) zout[(orow + r) * DYN + ocol] = acc[r];
}

// yt = z_next @ C^T + ut_dt @ D^T   (B x 50, tiny)
__global__ void k_yt(const float* __restrict__ znext, const float* __restrict__ utdt,
                     const float* __restrict__ C, const float* __restrict__ D,
                     float* __restrict__ yt) {
    int idx = blockIdx.x * 256 + threadIdx.x;
    if (idx >= BATCH * NOBS) return;
    int b = idx / NOBS, o = idx % NOBS;
    const float* z = znext + b * DYN;
    const float* c = C + o * DYN;
    float s = 0.f;
#pragma unroll 4
    for (int i = 0; i < DYN; ++i) s += z[i] * c[i];
    const float* u = utdt + b * UDIM;
    const float* d = D + o * UDIM;
#pragma unroll
    for (int i = 0; i < UDIM; ++i) s += u[i] * d[i];
    yt[idx] = s;
}

// ---------------------------------------------------------------------------
extern "C" void kernel_launch(void* const* d_in, const int* in_sizes, int n_in,
                              void* d_out, int out_size, void* d_ws, size_t ws_size,
                              hipStream_t stream) {
    const float* zd  = (const float*)d_in[0];
    const float* zs  = (const float*)d_in[1];
    const float* dt  = (const float*)d_in[2];
    const float* ut  = (const float*)d_in[3];
    const float* W1  = (const float*)d_in[4];
    const float* b1  = (const float*)d_in[5];
    const float* g1  = (const float*)d_in[6];
    const float* be1 = (const float*)d_in[7];
    const float* W2  = (const float*)d_in[8];
    const float* b2  = (const float*)d_in[9];
    const float* g2  = (const float*)d_in[10];
    const float* be2 = (const float*)d_in[11];
    const float* W3  = (const float*)d_in[12];
    const float* b3  = (const float*)d_in[13];
    const float* WK  = (const float*)d_in[14];
    const float* bK  = (const float*)d_in[15];
    const float* WL  = (const float*)d_in[16];
    const float* bL  = (const float*)d_in[17];
    const float* C   = (const float*)d_in[18];
    const float* D   = (const float*)d_in[19];

    float* z_out = (float*)d_out;                 // B x 128
    float* y_out = z_out + BATCH * DYN;           // B x 50

    // workspace carve-up (byte offsets, 256-aligned sizes)
    char* w = (char*)d_ws;
    auto carve = [&](size_t bytes) { char* p = w; w += (bytes + 255) & ~size_t(255); return p; };
    unsigned short* xb  = (unsigned short*)carve(BATCH * SELP * 2);
    unsigned short* w1p = (unsigned short*)carve(HID * SELP * 2);
    unsigned short* w2p = (unsigned short*)carve(HID * HID * 2);
    unsigned short* w3p = (unsigned short*)carve(HZ * HID * 2);
    float*          pre = (float*)carve(BATCH * HID * 4);   // reused 3x
    unsigned short* h1  = (unsigned short*)carve(BATCH * HID * 2);
    unsigned short* h2  = (unsigned short*)carve(BATCH * HID * 2);
    float*          hzx = (float*)carve(BATCH * HZP * 4);
    float*          udt = (float*)carve(BATCH * UDIM * 4);
    unsigned short* Aw  = (unsigned short*)carve((size_t)DYN * KBIG * 2);
    float*          sc1 = (float*)carve(HID * 4);
    float*          sh1 = (float*)carve(HID * 4);
    float*          sc2 = (float*)carve(HID * 4);
    float*          sh2 = (float*)carve(HID * 4);
    (void)ws_size; (void)in_sizes; (void)n_in; (void)out_size;

    // --- packing ---
    k_pack_x<<<(BATCH * SELP + 255) / 256, 256, 0, stream>>>(zd, zs, dt, xb);
    k_pack_w<<<(HID * SELP + 255) / 256, 256, 0, stream>>>(W1, w1p, DYN + STAT + 1, SELP, HID * SELP);
    k_pack_w<<<(HID * HID + 255) / 256, 256, 0, stream>>>(W2, w2p, HID, HID, HID * HID);
    k_pack_w<<<(HZ * HID + 255) / 256, 256, 0, stream>>>(W3, w3p, HID, HID, HZ * HID);
    k_pack_A<<<(DYN * KBIG + 255) / 256, 256, 0, stream>>>(WK, bK, WL, bL, Aw);
    k_utdt<<<(BATCH * UDIM + 255) / 256, 256, 0, stream>>>(ut, dt, udt);

    // --- selector MLP ---
    dim3 g1d(BATCH / 16, HID / 16);
    k_gemm<<<g1d, 32, 0, stream>>>(xb, w1p, b1, pre, SELP, HID);
    k_bnstats<<<HID, 256, 0, stream>>>(pre, g1, be1, sc1, sh1, HID);
    k_bnact<<<(BATCH * HID + 255) / 256, 256, 0, stream>>>(pre, sc1, sh1, h1, HID);

    k_gemm<<<g1d, 32, 0, stream>>>(h1, w2p, b2, pre, HID, HID);
    k_bnstats<<<HID, 256, 0, stream>>>(pre, g2, be2, sc2, sh2, HID);
    k_bnact<<<(BATCH * HID + 255) / 256, 256, 0, stream>>>(pre, sc2, sh2, h2, HID);

    dim3 g3d(BATCH / 16, HZ / 16);
    k_gemm<<<g3d, 32, 0, stream>>>(h2, w3p, b3, pre, HID, HZ);   // pre = hz (B x 192)
    k_hzx<<<(BATCH * HZP + 255) / 256, 256, 0, stream>>>(pre, hzx);

    // --- fused Koopman transition (the dominant GEMM) ---
    k_koopman<<<BATCH / 16, 256, 0, stream>>>(zd, udt, hzx, Aw, z_out);

    // --- observation ---
    k_yt<<<(BATCH * NOBS + 255) / 256, 256, 0, stream>>>(z_out, udt, C, D, y_out);
}
